// ConnectKeypointLayer_64372969832484
// MI455X (gfx1250) — matrix operand order repaired and checked by reading, not compiled, not run on hardware
//
#include <hip/hip_runtime.h>
#include <hip/hip_bf16.h>
#include <stdint.h>

#define NKP   21
#define CH    32
#define IMG   96
#define BATCH 8
#define NPIX  (BATCH*IMG*IMG)      // 73728
#define HALO  18
#define NHALO (HALO*HALO)          // 324
#define LDSPITCH 40                // f16 per pixel: 32 + 8 pad (80B = 20 banks -> conflict-free b128)

typedef __attribute__((ext_vector_type(16))) _Float16 v16h;
typedef __attribute__((ext_vector_type(8)))  float    v8f;

union AFrag { uint4 u[2]; v16h h; };

// fragment table ids in workspace
#define F_W2 0
#define F_WA 2                     // + n*2 + nh
#define F_WB 44                    // + (n*9+t)*2 + nh
#define NFRAG 422

__device__ __forceinline__ v16h wmma_load_frag(const uint32_t* fragbase, int fi, int lane) {
    const uint4* p = (const uint4*)(fragbase + (size_t)fi * 256 + lane * 8);
    AFrag af;
    af.u[0] = p[0];
    af.u[1] = p[1];
    return af.h;
}

__device__ __forceinline__ v8f wmma_f16(v16h a, v16h b, v8f c) {
    // D(f32 16x16) = A(f16 16x32) * B(f16 32x16) + C
    return __builtin_amdgcn_wmma_f32_16x16x32_f16(false, a, false, b, (short)0, c, false, false);
}

// XOR-butterfly add via DS_SWIZZLE group-of-32 mode:
// offset = xor_mask<<10 | or_mask<<5 | and_mask  (and_mask=0x1F keeps all lanes)
// masks 1,2,4,8 never cross the 16-lane pixel halves -> sums each half-group.
__device__ __forceinline__ float swz_add(float v, int imm_off) {
    return v; // overwritten by specializations below (kept for clarity)
}
__device__ __forceinline__ float xor16_sum(float v) {
    v += __int_as_float(__builtin_amdgcn_ds_swizzle(__float_as_int(v), 0x041F)); // xor 1
    v += __int_as_float(__builtin_amdgcn_ds_swizzle(__float_as_int(v), 0x081F)); // xor 2
    v += __int_as_float(__builtin_amdgcn_ds_swizzle(__float_as_int(v), 0x101F)); // xor 4
    v += __int_as_float(__builtin_amdgcn_ds_swizzle(__float_as_int(v), 0x201F)); // xor 8
    return v;   // every lane in its 16-lane half holds the half-group sum
}

// ---------------------------------------------------------------------------
// Kernel 0: pack 32x32 fp32 weight matrices into WMMA B-fragment layout (f16)
// B frag (32Kx16N f16): lane L -> N = L%16, kb = (L<16)?0:8; dword i holds
// K = kb+2i+(i>=4?8:0) (lo half) and +1 (hi half). One block (1 wave) per fragment.
// Blocks 422..442 compute wsum[n][c] = sum_d Wp[n][c][d].
// ---------------------------------------------------------------------------
__global__ void ckl_pack_kernel(const float* __restrict__ w2,
                                const float* __restrict__ wa,
                                const float* __restrict__ wb,
                                const float* __restrict__ Wp,
                                uint32_t* __restrict__ frag,
                                float* __restrict__ wsum) {
    int bid  = blockIdx.x;
    int lane = threadIdx.x;
    if (bid < NFRAG) {
        const float* src;
        int nh;
        if (bid < F_WA) {                       // w2
            src = w2; nh = bid;
        } else if (bid < F_WB) {                // wa[n]
            int id = bid - F_WA;
            src = wa + (size_t)(id >> 1) * 1024;
            nh = id & 1;
        } else {                                // wb[n][tap]
            int id = bid - F_WB;
            src = wb + (size_t)(id >> 1) * 1024;
            nh = id & 1;
        }
        int Nl = nh * 16 + (lane & 15);
        int kb = (lane < 16) ? 0 : 8;
        uint32_t* dst = frag + (size_t)bid * 256 + lane * 8;
#pragma unroll
        for (int i = 0; i < 8; ++i) {
            int k0 = kb + 2 * i + (i >= 4 ? 8 : 0);
            union { _Float16 h[2]; uint32_t u; } p;
            p.h[0] = (_Float16)src[k0 * 32 + Nl];
            p.h[1] = (_Float16)src[(k0 + 1) * 32 + Nl];
            dst[i] = p.u;
        }
    } else {
        int n = bid - NFRAG;                    // 0..20
        const float* wp = Wp + (size_t)n * 1024 + lane * 32;  // Wp[n][c=lane][:]
        float s = 0.f;
#pragma unroll
        for (int d = 0; d < 32; ++d) s += wp[d];
        wsum[n * 32 + lane] = s;
    }
}

// ---------------------------------------------------------------------------
// Kernel 1: f = relu(x @ w2 + b2) -> f16 [NPIX][32]; fsum = f.sum(ch) -> f32.
// One wave per 16-pixel M-tile; 8 waves / 256-thread block.
// ---------------------------------------------------------------------------
__global__ void __launch_bounds__(256)
ckl_f_kernel(const float* __restrict__ x,
             const uint32_t* __restrict__ frag,
             const float* __restrict__ b2,
             _Float16* __restrict__ fout,
             float* __restrict__ fsum) {
    int lane = threadIdx.x & 31;
    int wave = threadIdx.x >> 5;
    int tile = blockIdx.x * 8 + wave;           // 0..4607
    int m    = lane & 15;
    int kb   = (lane < 16) ? 0 : 8;             // A-frag K base
    int pA   = tile * 16 + m;

    // A fragment: row M = pixel pA, K = channels kb..kb+7, kb+16..kb+23
    const float* px = x + (size_t)pA * 32;
    v16h a;
#pragma unroll
    for (int i = 0; i < 8; ++i) a[i]     = (_Float16)px[kb + i];
#pragma unroll
    for (int i = 0; i < 8; ++i) a[8 + i] = (_Float16)px[kb + 16 + i];

    float pacc[8];
#pragma unroll
    for (int r = 0; r < 8; ++r) pacc[r] = 0.f;

    int mhi = (lane >= 16) ? 8 : 0;
#pragma unroll
    for (int nh = 0; nh < 2; ++nh) {
        v16h bw = wmma_load_frag(frag, F_W2 + nh, lane);
        v8f c = {};
        c = wmma_f16(a, bw, c);
        float bias = b2[nh * 16 + m];
#pragma unroll
        for (int r = 0; r < 8; ++r) {
            float v = c[r] + bias;
            v = v > 0.f ? v : 0.f;
            int pr = tile * 16 + r + mhi;       // pixel for this C element
            fout[(size_t)pr * 32 + nh * 16 + m] = (_Float16)v;
            pacc[r] += v;
        }
    }
#pragma unroll
    for (int r = 0; r < 8; ++r) {
        float v = xor16_sum(pacc[r]);
        if (m == 0) fsum[tile * 16 + r + mhi] = v;
    }
}

// ---------------------------------------------------------------------------
// Kernel 2 (fused): for one (16x16 tile, batch image, keypoint):
//   stage B: h = relu(f @ wa_n + ba_n) on 18x18 halo -> LDS (f16, padded pitch)
//   stage C: 3x3 implicit-GEMM conv from LDS + relu + channel-dot(wsum) +
//            fsum -> out[B,H,W,21]
// ---------------------------------------------------------------------------
__global__ void __launch_bounds__(256)
ckl_conv_kernel(const _Float16* __restrict__ f,
                const float* __restrict__ fsum,
                const uint32_t* __restrict__ frag,
                const float* __restrict__ wsum,
                const float* __restrict__ ba,
                const float* __restrict__ bb,
                float* __restrict__ out) {
    __shared__ __align__(16) _Float16 lds_h[NHALO * LDSPITCH];  // 25.9 KB

    int lane = threadIdx.x & 31;
    int wave = threadIdx.x >> 5;
    int tx = blockIdx.x % 6, ty = blockIdx.x / 6;
    int bimg = blockIdx.y;
    int n    = blockIdx.z;
    int x0 = tx * 16, y0 = ty * 16;
    int m   = lane & 15;
    int kb  = (lane < 16) ? 0 : 8;
    int mhi = (lane >= 16) ? 8 : 0;

    // ---- stage B: build h on the halo ------------------------------------
    v16h bwa0 = wmma_load_frag(frag, F_WA + n * 2 + 0, lane);
    v16h bwa1 = wmma_load_frag(frag, F_WA + n * 2 + 1, lane);
    float ba0 = ba[n * 32 + m];
    float ba1 = ba[n * 32 + 16 + m];

    uint4 z4; z4.x = z4.y = z4.z = z4.w = 0u;
    for (int tile = wave; tile < 21; tile += 8) {   // ceil(324/16) M-tiles
        int hpA = tile * 16 + m;
        AFrag af; af.u[0] = z4; af.u[1] = z4;
        if (hpA < NHALO) {
            int hy = hpA / HALO, hx = hpA % HALO;
            int gy = y0 - 1 + hy, gx = x0 - 1 + hx;
            if (gy >= 0 && gy < IMG && gx >= 0 && gx < IMG) {
                const _Float16* fp = f + ((size_t)((bimg * IMG + gy) * IMG + gx)) * 32;
                af.u[0] = *(const uint4*)(fp + kb);
                af.u[1] = *(const uint4*)(fp + kb + 16);
            }
        }
#pragma unroll
        for (int nh = 0; nh < 2; ++nh) {
            v8f c = {};
            c = wmma_f16(af.h, nh ? bwa1 : bwa0, c);
            float bias = nh ? ba1 : ba0;
#pragma unroll
            for (int r = 0; r < 8; ++r) {
                int hps = tile * 16 + r + mhi;
                if (hps < NHALO) {
                    int hy = hps / HALO, hx = hps % HALO;
                    int gy = y0 - 1 + hy, gx = x0 - 1 + hx;
                    bool valid = (gy >= 0 && gy < IMG && gx >= 0 && gx < IMG);
                    float v = c[r] + bias;
                    v = v > 0.f ? v : 0.f;
                    lds_h[hps * LDSPITCH + nh * 16 + m] =
                        valid ? (_Float16)v : (_Float16)0.f;
                }
            }
        }
    }
    __syncthreads();

    // ---- stage C: 3x3 conv as 9 WMMAs per (row, N-half) -------------------
    float pacc[2][8];
#pragma unroll
    for (int rr = 0; rr < 2; ++rr)
#pragma unroll
        for (int r = 0; r < 8; ++r) pacc[rr][r] = 0.f;

#pragma unroll
    for (int nh = 0; nh < 2; ++nh) {
        v16h bfr[9];
#pragma unroll
        for (int t = 0; t < 9; ++t)
            bfr[t] = wmma_load_frag(frag, F_WB + (n * 9 + t) * 2 + nh, lane);
        float bias = bb[n * 32 + nh * 16 + m];
        float wsv  = wsum[n * 32 + nh * 16 + m];
#pragma unroll
        for (int rr = 0; rr < 2; ++rr) {
            int yy = wave + rr * 8;             // output row in tile
            v8f c = {};
#pragma unroll
            for (int t = 0; t < 9; ++t) {
                int dy = t / 3 - 1, dx = t % 3 - 1;
                int pix = (yy + 1 + dy) * HALO + (m + 1 + dx);
                AFrag af;
                af.u[0] = *(const uint4*)&lds_h[pix * LDSPITCH + kb];
                af.u[1] = *(const uint4*)&lds_h[pix * LDSPITCH + kb + 16];
                c = wmma_f16(af.h, bfr[t], c);
            }
#pragma unroll
            for (int r = 0; r < 8; ++r) {
                float v = c[r] + bias;
                v = v > 0.f ? v : 0.f;
                pacc[rr][r] += v * wsv;
            }
        }
    }

    // ---- reduce over channels, add fsum, write [B,H,W,21] -----------------
#pragma unroll
    for (int rr = 0; rr < 2; ++rr) {
        int yy = wave + rr * 8;
#pragma unroll
        for (int r = 0; r < 8; ++r) {
            float v = xor16_sum(pacc[rr][r]);
            if (m == 0) {
                int gx = x0 + r + mhi;
                int gy = y0 + yy;
                int p  = (bimg * IMG + gy) * IMG + gx;
                out[(size_t)p * NKP + n] = fsum[p] + v;
            }
        }
    }
}

extern "C" void kernel_launch(void* const* d_in, const int* in_sizes, int n_in,
                              void* d_out, int out_size, void* d_ws, size_t ws_size,
                              hipStream_t stream) {
    (void)in_sizes; (void)n_in; (void)out_size; (void)ws_size;
    const float* x  = (const float*)d_in[0];
    const float* w2 = (const float*)d_in[1];
    const float* b2 = (const float*)d_in[2];
    const float* wa = (const float*)d_in[3];
    const float* ba = (const float*)d_in[4];
    const float* wb = (const float*)d_in[5];
    const float* bb = (const float*)d_in[6];
    const float* Wp = (const float*)d_in[7];
    float* out = (float*)d_out;

    char* ws = (char*)d_ws;
    _Float16* fbuf = (_Float16*)(ws);                              // 4,718,592 B
    float*    fsum = (float*)(ws + 4718592);                       //   294,912 B
    uint32_t* frag = (uint32_t*)(ws + 4718592 + 294912);           //   432,128 B
    float*    wsum = (float*)(ws + 4718592 + 294912 + 432128);     //     2,688 B

    ckl_pack_kernel<<<NFRAG + NKP, 32, 0, stream>>>(w2, wa, wb, Wp, frag, wsum);
    ckl_f_kernel<<<NPIX / 16 / 8, 256, 0, stream>>>(x, frag, b2, fbuf, fsum);
    ckl_conv_kernel<<<dim3(36, BATCH, NKP), 256, 0, stream>>>(fbuf, fsum, frag,
                                                              wsum, ba, bb, out);
}
